// TransBasicBlockSparse_61976378081615
// MI455X (gfx1250) — matrix operand order repaired and checked by reading, not compiled, not run on hardware
//
#include <hip/hip_runtime.h>
#include <hip/hip_bf16.h>

// ---------------------------------------------------------------------------
// Sparse submanifold ResNet block for MI455X (gfx1250, wave32, WMMA).
// bf16 activations (fit in 192MB L2), v_wmma_f32_16x16x32_bf16 with fp32
// accumulation, async global->LDS double-buffered gather (ASYNCcnt),
// BN stats fused into the conv epilogue.
// ---------------------------------------------------------------------------

typedef __attribute__((ext_vector_type(16))) __bf16 bf16x16;
typedef __attribute__((ext_vector_type(8)))  __bf16 bf16x8;
typedef __attribute__((ext_vector_type(8)))  float  floatx8;

#define C_CH   128
#define KOFF   9
#define AST    136   // LDS row stride in bf16 (256B row + 16B pad -> bank shift)
#define EPSBN  1e-4f

static __host__ __device__ inline size_t align256(size_t x) { return (x + 255) & ~(size_t)255; }

// ---------------------------------------------------------------------------
// Kernel 1: cast x (fp32 [N,128]) -> bf16 [N+1,128], pad row N = 0
// ---------------------------------------------------------------------------
__global__ void cast_x_kernel(const float* __restrict__ x, __bf16* __restrict__ xbf, int Nsites)
{
    size_t valid = (size_t)Nsites * C_CH;
    size_t total = (size_t)(Nsites + 1) * C_CH;
    for (size_t i = (size_t)blockIdx.x * blockDim.x + threadIdx.x; i < total;
         i += (size_t)gridDim.x * blockDim.x) {
        xbf[i] = (i < valid) ? (__bf16)x[i] : (__bf16)0.0f;
    }
}

// ---------------------------------------------------------------------------
// Kernel 2: pack W [9,128,128] fp32 -> bf16 WMMA B-fragment blocks.
// Block = (k, kstep, ntile): 32 lanes x 16 bf16, contiguous per lane (32B).
// B pack order used here (contiguous k, as A): lane L holds column n = L&15; K element t maps to
// K = (L>>4)*16 + t (lanes 0-15: K=0..15, lanes 16-31: K=16..31).
// ---------------------------------------------------------------------------
__global__ void pack_w_kernel(const float* __restrict__ W, __bf16* __restrict__ pack)
{
    int idx = blockIdx.x * blockDim.x + threadIdx.x;
    if (idx >= KOFF * 4 * 8 * 512) return;   // 147456 elements
    int t    = idx & 15;
    int lane = (idx >> 4) & 31;
    int nt   = (idx >> 9) & 7;
    int ks   = (idx >> 12) & 3;
    int k    = idx >> 14;
    int half = lane >> 4;
    int n    = nt * 16 + (lane & 15);        // output channel (B column)
    int kk   = half * 16 + t;                // K within 32-wide step
    int c    = ks * 32 + kk;                 // input channel (B row)
    pack[idx] = (__bf16)W[((size_t)k * C_CH + c) * C_CH + n];
}

// ---------------------------------------------------------------------------
// Kernel 3/5: gather-GEMM.  64 sites/block, 8 waves; wave w owns output cols
// [16w,16w+16).  Per offset k: async gather of 64 neighbor rows into a
// double-buffered LDS tile (overlapped with WMMA compute of offset k-1),
// then 4 k-steps x 4 m-tiles of v_wmma_f32_16x16x32_bf16.
// Fused BN-stat reduction (sum, sumsq per channel) in the epilogue.
// ---------------------------------------------------------------------------
template <bool WRITE_BF16>
__global__ __launch_bounds__(256)
void conv_wmma_kernel(const __bf16* __restrict__ xin,    // [Nsites+1, 128] bf16
                      const int*    __restrict__ nbr,    // [Nsites, 9]
                      const __bf16* __restrict__ wpack,  // packed weights
                      void*         __restrict__ outp,   // bf16 or fp32 [Nsites,128]
                      float* __restrict__ gsum, float* __restrict__ gsq,
                      int Nsites)
{
    __shared__ __align__(16) __bf16 As[2][64 * AST];
    __shared__ float psum[8][32];
    __shared__ float psq[8][32];

    const int tid  = threadIdx.x;
    const int wv   = tid >> 5;
    const int lane = tid & 31;
    const int half = lane >> 4;
    const int l15  = lane & 15;
    const int m0   = blockIdx.x * 64;
    const int r    = tid >> 2;   // gather: row 0..63
    const int seg  = tid & 3;    // gather: 64B segment

    floatx8 acc[4] = {};

    // Issue one gather stage (offset kk) into LDS buffer `buf` using the
    // CDNA5 async global->LDS path (4 async instructions per wave, ASYNCcnt).
    auto issue_stage = [&](int kk, int buf) {
        int site = m0 + r;
        int g = (site < Nsites) ? nbr[site * KOFF + kk] : Nsites;  // sentinel -> zero row
        unsigned ldsb = (unsigned)(uintptr_t)(&As[buf][r * AST]) + (unsigned)(seg * 64);
        const char* srcb = (const char*)(xin + (size_t)g * C_CH) + seg * 64;
#pragma unroll
        for (int j = 0; j < 4; ++j) {
            asm volatile("global_load_async_to_lds_b128 %0, %1, off"
                         :: "v"(ldsb + (unsigned)(j * 16)), "v"(srcb + j * 16)
                         : "memory");
        }
    };

    issue_stage(0, 0);

    for (int k = 0; k < KOFF; ++k) {
        if (k + 1 < KOFF) {
            // prefetch next offset's rows into the other buffer (overlaps WMMA)
            issue_stage(k + 1, (k + 1) & 1);
            // async loads complete in order: cnt<=4 => stage k's 4 loads landed
            asm volatile("s_wait_asynccnt 0x4" ::: "memory");
        } else {
            asm volatile("s_wait_asynccnt 0x0" ::: "memory");
        }
        __syncthreads();   // all waves' stage-k data visible in LDS

        const __bf16* Ac = &As[k & 1][0];
#pragma unroll
        for (int ks = 0; ks < 4; ++ks) {
            // B fragment: one 32B contiguous chunk per lane
            const bf16x8* wp =
                (const bf16x8*)(wpack + (((size_t)(k * 4 + ks) * 8 + wv) * 512) + lane * 16);
            union { bf16x16 v; bf16x8 h[2]; } ub;
            ub.h[0] = wp[0];
            ub.h[1] = wp[1];

#pragma unroll
            for (int mt = 0; mt < 4; ++mt) {
                // A fragment from LDS, contiguous k order (same as B pack):
                // lane half 0: K = 0..15; half 1: K = 16..31
                const __bf16* ap = Ac + (mt * 16 + l15) * AST + ks * 32 + half * 16;
                union { bf16x16 v; bf16x8 h[2]; } ua;
                ua.h[0] = *(const bf16x8*)ap;
                ua.h[1] = *(const bf16x8*)(ap + 8);
                acc[mt] = __builtin_amdgcn_wmma_f32_16x16x32_bf16(
                    false, ua.v, false, ub.v, (short)0, acc[mt], false, false);
            }
        }
        __syncthreads();   // protect the buffer that the next issue overwrites
    }

    // ---- epilogue: store + BN statistics ----
    const int col = wv * 16 + l15;
    float s = 0.f, q = 0.f;
#pragma unroll
    for (int mt = 0; mt < 4; ++mt) {
#pragma unroll
        for (int v = 0; v < 8; ++v) {
            int row = m0 + mt * 16 + half * 8 + v;   // D layout: M = v + 8*half
            float val = acc[mt][v];
            if (row < Nsites) {
                s += val;
                q += val * val;
                if (WRITE_BF16)
                    ((__bf16*)outp)[(size_t)row * C_CH + col] = (__bf16)val;
                else
                    ((float*)outp)[(size_t)row * C_CH + col] = val;
            }
        }
    }
    psum[wv][lane] = s;
    psq[wv][lane]  = q;
    __syncthreads();

    if (tid < C_CH) {
        int c = tid;
        float ts = psum[c >> 4][c & 15] + psum[c >> 4][(c & 15) + 16];
        float tq = psq[c >> 4][c & 15] + psq[c >> 4][(c & 15) + 16];
        atomicAdd(&gsum[c], ts);
        atomicAdd(&gsq[c], tq);
    }
}

// ---------------------------------------------------------------------------
// Kernel 4/6: BN finalize: (sum, sumsq) -> per-channel scale/shift
// ---------------------------------------------------------------------------
__global__ void finalize_stats_kernel(const float* __restrict__ gsum, const float* __restrict__ gsq,
                                      const float* __restrict__ gamma, const float* __restrict__ beta,
                                      float* __restrict__ scale, float* __restrict__ shift, int Nsites)
{
    int c = threadIdx.x;
    if (c < C_CH) {
        float inv = 1.0f / (float)Nsites;
        float mu  = gsum[c] * inv;
        float var = gsq[c] * inv - mu * mu;
        float sc  = gamma[c] * rsqrtf(var + EPSBN);
        scale[c] = sc;
        shift[c] = beta[c] - mu * sc;
    }
}

// ---------------------------------------------------------------------------
// Kernel: bn1 + relu in place on bf16 h1 (also rewrites pad row to 0)
// ---------------------------------------------------------------------------
__global__ void bn_relu_bf16_kernel(__bf16* __restrict__ h,
                                    const float* __restrict__ scale,
                                    const float* __restrict__ shift, int Nsites)
{
    size_t valid = (size_t)Nsites * C_CH;
    size_t total = (size_t)(Nsites + 1) * C_CH;
    for (size_t i = (size_t)blockIdx.x * blockDim.x + threadIdx.x; i < total;
         i += (size_t)gridDim.x * blockDim.x) {
        if (i < valid) {
            int c = (int)(i & (C_CH - 1));
            float v = (float)h[i] * scale[c] + shift[c];
            h[i] = (__bf16)fmaxf(v, 0.0f);
        } else {
            h[i] = (__bf16)0.0f;
        }
    }
}

// ---------------------------------------------------------------------------
// Kernel: bn2 + residual + relu, in place on fp32 d_out
// ---------------------------------------------------------------------------
__global__ void bn_add_relu_kernel(float* __restrict__ out, const float* __restrict__ x,
                                   const float* __restrict__ scale,
                                   const float* __restrict__ shift, int Nsites)
{
    size_t total = (size_t)Nsites * C_CH;
    for (size_t i = (size_t)blockIdx.x * blockDim.x + threadIdx.x; i < total;
         i += (size_t)gridDim.x * blockDim.x) {
        int c = (int)(i & (C_CH - 1));
        float v = out[i] * scale[c] + shift[c] + x[i];
        out[i] = fmaxf(v, 0.0f);
    }
}

// ---------------------------------------------------------------------------
// Host side
// ---------------------------------------------------------------------------
extern "C" void kernel_launch(void* const* d_in, const int* in_sizes, int n_in,
                              void* d_out, int out_size, void* d_ws, size_t ws_size,
                              hipStream_t stream)
{
    const float* x      = (const float*)d_in[0];
    const float* W1     = (const float*)d_in[1];
    const float* W2     = (const float*)d_in[2];
    const float* gamma1 = (const float*)d_in[3];
    const float* beta1  = (const float*)d_in[4];
    const float* gamma2 = (const float*)d_in[5];
    const float* beta2  = (const float*)d_in[6];
    const int*   nbr    = (const int*)d_in[7];
    const int Nsites = in_sizes[0] / C_CH;

    // Workspace layout
    char* ws = (char*)d_ws;
    const size_t szAct  = align256((size_t)(Nsites + 1) * C_CH * sizeof(__bf16)); // ~128MB
    const size_t szPack = align256((size_t)KOFF * 4 * 8 * 512 * sizeof(__bf16));  // 288KB

    __bf16* xbf = (__bf16*)ws;                ws += szAct;
    __bf16* h1  = (__bf16*)ws;                ws += szAct;
    __bf16* p1  = (__bf16*)ws;                ws += szPack;
    __bf16* p2  = (__bf16*)ws;                ws += szPack;
    float*  st  = (float*)ws;
    // st: [0]=sum1 [128]=sq1 [256]=sum2 [384]=sq2 [512]=scale1 [640]=shift1 [768]=scale2 [896]=shift2

    // zero BN accumulators (graph-capture safe)
    hipMemsetAsync(st, 0, 512 * sizeof(float), stream);

    // casts / weight packing
    cast_x_kernel<<<8192, 256, 0, stream>>>(x, xbf, Nsites);
    pack_w_kernel<<<(KOFF * 4 * 8 * 512 + 255) / 256, 256, 0, stream>>>(W1, p1);
    pack_w_kernel<<<(KOFF * 4 * 8 * 512 + 255) / 256, 256, 0, stream>>>(W2, p2);

    const int nblk = (Nsites + 63) / 64;

    // conv1 (-> bf16 pre-BN h1) + stats1
    conv_wmma_kernel<true><<<nblk, 256, 0, stream>>>(xbf, nbr, p1, (void*)h1,
                                                     st + 0, st + 128, Nsites);
    finalize_stats_kernel<<<1, 128, 0, stream>>>(st + 0, st + 128, gamma1, beta1,
                                                 st + 512, st + 640, Nsites);
    bn_relu_bf16_kernel<<<8192, 256, 0, stream>>>(h1, st + 512, st + 640, Nsites);

    // conv2 (-> fp32 pre-BN directly into d_out) + stats2
    conv_wmma_kernel<false><<<nblk, 256, 0, stream>>>(h1, nbr, p2, d_out,
                                                      st + 256, st + 384, Nsites);
    finalize_stats_kernel<<<1, 128, 0, stream>>>(st + 256, st + 384, gamma2, beta2,
                                                 st + 768, st + 896, Nsites);
    bn_add_relu_kernel<<<8192, 256, 0, stream>>>((float*)d_out, x, st + 768, st + 896, Nsites);

    (void)n_in; (void)out_size; (void)ws_size;
}
